// hyperNet_25915832664751
// MI455X (gfx1250) — matrix operand order repaired and checked by reading, not compiled, not run on hardware
//
#include <hip/hip_runtime.h>
#include <hip/hip_bf16.h>

// ---------------------------------------------------------------------------
// hyperNet on MI455X (gfx1250).
// All GEMM-shaped work -> v_wmma_f32_16x16x32_bf16 (bf16 in, f32 accum).
// Every WMMA operand is K-contiguous in memory => 128-bit vector loads:
//   - L is symmetric, so @L and @L^T are both C = A * B^T (NT GEMM).
//   - Gram Zc^T Zc = NT GEMM on pre-transposed ZcT.
//   - Conv weights repacked (9,M,K) k-innermost; conv inputs stored (N,C).
// Each wave computes a 32x32 output tile: 2 A-frags x 2 B-frags -> 4 WMMAs
// per k-step on 4 independent accumulator chains (hides XDL latency).
// ---------------------------------------------------------------------------

typedef __bf16 v16bf  __attribute__((ext_vector_type(16)));
typedef __bf16 bf16x8 __attribute__((ext_vector_type(8)));
typedef float  v8f    __attribute__((ext_vector_type(8)));

#define NN      1024
#define NOPEN   128
#define NHID    256
#define NSTART  40
#define NCLOSE  3
#define NLAYERS 40
#define H2      0.01f
#define EPS_IN  1e-5f

#define WMMA_BF16(a, b, c) \
  __builtin_amdgcn_wmma_f32_16x16x32_bf16(false, (a), false, (b), (short)0, (c), false, false)

__device__ __forceinline__ v16bf cat8(bf16x8 lo, bf16x8 hi) {
  return __builtin_shufflevector(lo, hi, 0, 1, 2, 3, 4, 5, 6, 7,
                                 8, 9, 10, 11, 12, 13, 14, 15);
}

// Fragment layouts (wave32):
//  A 16x32: lane L: m=L&15, kb=(L>>4)*16 -> K = kb+{0..15}   (contiguous!)
//  B 32x16: lane L: n=L&15, kb=(L>>4)*16 -> K = kb+{0..15}   (contiguous!)
//  C 16x16: lane L: n=L&15, VGPR v -> m = v + (L>>4)*8.

// ---------------- NT GEMM: C(f32, MxN) = A(MxK) * B(NxK)^T -----------------
// One wave per 32x32 output tile.
__global__ __launch_bounds__(32) void k_gemm_nt(
    const __bf16* __restrict__ A, int lda,
    const __bf16* __restrict__ B, int ldb,
    float* __restrict__ C, int ldc, int K) {
  const int lane = threadIdx.x;
  const int n0 = blockIdx.x * 32;
  const int m0 = blockIdx.y * 32;
  const int bn0 = n0 + (lane & 15);
  const int akb = (lane >> 4) << 4;
  const int bkb = (lane >> 4) << 4;
  const __bf16* Ar0 = A + (size_t)(m0 + (lane & 15)) * lda;
  const __bf16* Ar1 = Ar0 + (size_t)16 * lda;
  const __bf16* Br0 = B + (size_t)bn0 * ldb;
  const __bf16* Br1 = Br0 + (size_t)16 * ldb;
  v8f acc00 = {}, acc01 = {}, acc10 = {}, acc11 = {};
  for (int kk = 0; kk < K; kk += 32) {
    v16bf a0 = cat8(*(const bf16x8*)(Ar0 + kk + akb),
                    *(const bf16x8*)(Ar0 + kk + akb + 8));
    v16bf a1 = cat8(*(const bf16x8*)(Ar1 + kk + akb),
                    *(const bf16x8*)(Ar1 + kk + akb + 8));
    v16bf b0 = cat8(*(const bf16x8*)(Br0 + kk + bkb),
                    *(const bf16x8*)(Br0 + kk + bkb + 8));
    v16bf b1 = cat8(*(const bf16x8*)(Br1 + kk + bkb),
                    *(const bf16x8*)(Br1 + kk + bkb + 8));
    acc00 = WMMA_BF16(a0, b0, acc00);
    acc01 = WMMA_BF16(a0, b1, acc01);
    acc10 = WMMA_BF16(a1, b0, acc10);
    acc11 = WMMA_BF16(a1, b1, acc11);
  }
#pragma unroll
  for (int v = 0; v < 8; ++v) {
    const int m = m0 + v + ((lane >> 4) << 3);
    C[(size_t)m * ldc + bn0]             = acc00[v];
    C[(size_t)m * ldc + bn0 + 16]        = acc01[v];
    C[(size_t)(m + 16) * ldc + bn0]      = acc10[v];
    C[(size_t)(m + 16) * ldc + bn0 + 16] = acc11[v];
  }
}

// ---------------- Conv1d / Conv1d^T as implicit GEMM ------------------------
// Wp: (9, M, Kc) bf16, K innermost.  XT: (NN, Kc) bf16 (input transposed).
// forward: shift = t-4, transposed: shift = 4-t.  Out-of-range cols -> 0.
__global__ __launch_bounds__(32) void k_conv_nt(
    const __bf16* __restrict__ Wp, const __bf16* __restrict__ XT,
    float* __restrict__ Yf, __bf16* __restrict__ Yb,
    int M, int Kc, int transposed) {
  const int lane = threadIdx.x;
  const int n0 = blockIdx.x * 32;
  const int m0 = blockIdx.y * 32;
  const int bn0 = n0 + (lane & 15);
  const int akb = (lane >> 4) << 4;
  const int bkb = (lane >> 4) << 4;
  const __bf16* Ar0 = Wp + (size_t)(m0 + (lane & 15)) * Kc;
  const __bf16* Ar1 = Ar0 + (size_t)16 * Kc;
  const size_t wstride = (size_t)M * Kc;
  v8f acc00 = {}, acc01 = {}, acc10 = {}, acc11 = {};
  const bf16x8 zz = {};
  for (int t = 0; t < 9; ++t) {
    const int shift = transposed ? (4 - t) : (t - 4);
    const int col0 = bn0 + shift, col1 = col0 + 16;
    const bool ok0 = (unsigned)col0 < NN, ok1 = (unsigned)col1 < NN;
    const __bf16* W0_ = Ar0 + (size_t)t * wstride;
    const __bf16* W1_ = Ar1 + (size_t)t * wstride;
    const __bf16* X0 = XT + (size_t)(ok0 ? col0 : 0) * Kc;
    const __bf16* X1 = XT + (size_t)(ok1 ? col1 : 0) * Kc;
    for (int kk = 0; kk < Kc; kk += 32) {
      v16bf a0 = cat8(*(const bf16x8*)(W0_ + kk + akb),
                      *(const bf16x8*)(W0_ + kk + akb + 8));
      v16bf a1 = cat8(*(const bf16x8*)(W1_ + kk + akb),
                      *(const bf16x8*)(W1_ + kk + akb + 8));
      bf16x8 b0l = *(const bf16x8*)(X0 + kk + bkb);
      bf16x8 b0h = *(const bf16x8*)(X0 + kk + bkb + 8);
      bf16x8 b1l = *(const bf16x8*)(X1 + kk + bkb);
      bf16x8 b1h = *(const bf16x8*)(X1 + kk + bkb + 8);
      if (!ok0) { b0l = zz; b0h = zz; }
      if (!ok1) { b1l = zz; b1h = zz; }
      const v16bf b0 = cat8(b0l, b0h);
      const v16bf b1 = cat8(b1l, b1h);
      acc00 = WMMA_BF16(a0, b0, acc00);
      acc01 = WMMA_BF16(a0, b1, acc01);
      acc10 = WMMA_BF16(a1, b0, acc10);
      acc11 = WMMA_BF16(a1, b1, acc11);
    }
  }
#pragma unroll
  for (int v = 0; v < 8; ++v) {
    const int m = m0 + v + ((lane >> 4) << 3);
    if (Yf) {
      Yf[(size_t)m * NN + bn0]             = acc00[v];
      Yf[(size_t)m * NN + bn0 + 16]        = acc01[v];
      Yf[(size_t)(m + 16) * NN + bn0]      = acc10[v];
      Yf[(size_t)(m + 16) * NN + bn0 + 16] = acc11[v];
    }
    if (Yb) {
      Yb[(size_t)m * NN + bn0]             = (__bf16)acc00[v];
      Yb[(size_t)m * NN + bn0 + 16]        = (__bf16)acc01[v];
      Yb[(size_t)(m + 16) * NN + bn0]      = (__bf16)acc10[v];
      Yb[(size_t)(m + 16) * NN + bn0 + 16] = (__bf16)acc11[v];
    }
  }
}

// ---------------- Per-layer weight repack to bf16, K innermost --------------
// Wf[b]: (9,256,128)  Wf[t][co][ci] = W[b][(co*128+ci)*9+t]
// Wt[b]: (9,128,256)  Wt[t][ci][co] = W[b][(co*128+ci)*9+t]
__global__ __launch_bounds__(256) void k_repack(
    const float* __restrict__ W0, const float* __restrict__ W1,
    __bf16* __restrict__ Wf0, __bf16* __restrict__ Wf1,
    __bf16* __restrict__ Wt0, __bf16* __restrict__ Wt1) {
  const int idx = blockIdx.x * 256 + threadIdx.x;   // 9*256*128
  if (idx >= 9 * NHID * NOPEN) return;
  const int t  = idx / (NHID * NOPEN);
  const int r  = idx % (NHID * NOPEN);
  {
    const int co = r / NOPEN, ci = r % NOPEN;
    const int src = (co * NOPEN + ci) * 9 + t;
    Wf0[idx] = (__bf16)W0[src];
    Wf1[idx] = (__bf16)W1[src];
  }
  {
    const int ci = r / NHID, co = r % NHID;
    const int src = (co * NOPEN + ci) * 9 + t;
    Wt0[idx] = (__bf16)W0[src];
    Wt1[idx] = (__bf16)W1[src];
  }
}

// ---------------- Instance norm (biased) + ReLU -> bf16, TRANSPOSED out -----
__global__ __launch_bounds__(256) void k_instnorm_relu_T(
    const float* __restrict__ Y, __bf16* __restrict__ OutT, int Mrows) {
  __shared__ float s1[256], s2[256];
  const int r = blockIdx.x, tid = threadIdx.x;
  const float* y = Y + (size_t)r * NN;
  float a = 0.f, b = 0.f;
  for (int j = tid; j < NN; j += 256) { float v = y[j]; a += v; b += v * v; }
  s1[tid] = a; s2[tid] = b;
  __syncthreads();
  for (int st = 128; st > 0; st >>= 1) {
    if (tid < st) { s1[tid] += s1[tid + st]; s2[tid] += s2[tid + st]; }
    __syncthreads();
  }
  const float mean = s1[0] * (1.0f / NN);
  const float var  = s2[0] * (1.0f / NN) - mean * mean;
  const float inv  = rsqrtf(var + EPS_IN);
  for (int j = tid; j < NN; j += 256) {
    float v = (y[j] - mean) * inv;
    OutT[(size_t)j * Mrows + r] = (__bf16)(v > 0.f ? v : 0.f);
  }
}

// ---------------- Graph Laplacian pieces ------------------------------------
__global__ __launch_bounds__(256) void k_transp_zc(const float* __restrict__ Zc,
                                                   __bf16* __restrict__ ZcT) {
  const int idx = blockIdx.x * 256 + threadIdx.x;   // 1024*128
  const int n = idx / NOPEN, c = idx % NOPEN;
  ZcT[idx] = (__bf16)Zc[(size_t)c * NN + n];
}

__global__ __launch_bounds__(128) void k_colsq(const float* __restrict__ Zc,
                                               float* __restrict__ sq) {
  __shared__ float s[128];
  const int col = blockIdx.x, tid = threadIdx.x;
  const float v = Zc[(size_t)tid * NN + col];
  s[tid] = v * v;
  __syncthreads();
  for (int st = 64; st > 0; st >>= 1) {
    if (tid < st) s[tid] += s[tid + st];
    __syncthreads();
  }
  if (tid == 0) sq[col] = s[0];
}

__global__ __launch_bounds__(256) void k_d2row(const float* __restrict__ G,
                                               const float* __restrict__ sq,
                                               float* __restrict__ D2,
                                               float* __restrict__ partial) {
  __shared__ float s[256];
  const int i = blockIdx.x, tid = threadIdx.x;
  const float sqi = sq[i];
  float acc = 0.f;
  for (int j = tid; j < NN; j += 256) {
    float d = sqi + sq[j] - 2.0f * G[(size_t)i * NN + j];
    d = d > 0.f ? d : 0.f;
    D2[(size_t)i * NN + j] = d;
    acc += d;
  }
  s[tid] = acc;
  __syncthreads();
  for (int st = 128; st > 0; st >>= 1) {
    if (tid < st) s[tid] += s[tid + st];
    __syncthreads();
  }
  if (tid == 0) partial[i] = s[0];
}

__global__ __launch_bounds__(256) void k_sigma(const float* __restrict__ partial,
                                               float* __restrict__ sigma) {
  __shared__ float s[256];
  const int tid = threadIdx.x;
  float acc = 0.f;
  for (int j = tid; j < NN; j += 256) acc += partial[j];
  s[tid] = acc;
  __syncthreads();
  for (int st = 128; st > 0; st >>= 1) {
    if (tid < st) s[tid] += s[tid + st];
    __syncthreads();
  }
  if (tid == 0) sigma[0] = s[0] * (1.0f / ((float)NN * (float)NN)) + 1e-12f;
}

__global__ __launch_bounds__(256) void k_wgdeg(float* __restrict__ D2_Wg,
                                               const float* __restrict__ sigma,
                                               float* __restrict__ deg) {
  __shared__ float s[256];
  const int i = blockIdx.x, tid = threadIdx.x;
  const float inv_sig = 1.0f / sigma[0];
  float acc = 0.f;
  for (int j = tid; j < NN; j += 256) {
    const float w = __expf(-D2_Wg[(size_t)i * NN + j] * inv_sig);
    D2_Wg[(size_t)i * NN + j] = w;   // Wg overwrites D2
    acc += w;
  }
  s[tid] = acc;
  __syncthreads();
  for (int st = 128; st > 0; st >>= 1) {
    if (tid < st) s[tid] += s[tid + st];
    __syncthreads();
  }
  if (tid == 0) deg[i] = s[0];
}

__global__ __launch_bounds__(256) void k_lap(const float* __restrict__ Wg,
                                             const float* __restrict__ deg,
                                             __bf16* __restrict__ Lb) {
  const int idx = blockIdx.x * 256 + threadIdx.x;   // 1M
  const int i = idx >> 10, j = idx & (NN - 1);
  const float di = deg[i], dj = deg[j];
  const float v = ((i == j) ? di : 0.f) - Wg[idx];
  Lb[idx] = (__bf16)(rsqrtf(di) * v * rsqrtf(dj));
}

// ---------------- Small elementwise / projection kernels --------------------
__global__ __launch_bounds__(256) void k_open(const float* __restrict__ Kopen,
                                              const float* __restrict__ Z,
                                              float* __restrict__ Zc,
                                              float* __restrict__ Zold) {
  const int idx = blockIdx.x * 256 + threadIdx.x;   // 128*1024
  const int r = idx >> 10, c = idx & (NN - 1);
  float acc = 0.f;
#pragma unroll
  for (int k = 0; k < NSTART; ++k) acc += Kopen[r * NSTART + k] * Z[k * NN + c];
  Zc[idx] = acc;
  Zold[idx] = acc;
}

__global__ __launch_bounds__(256) void k_biasadd_T(const float* __restrict__ Zc,
                                                   const float* __restrict__ BiasL,
                                                   __bf16* __restrict__ XT0,
                                                   __bf16* __restrict__ XT1) {
  const int idx = blockIdx.x * 256 + threadIdx.x;   // 1024*128
  const int n = idx / NOPEN, c = idx % NOPEN;
  const float z = Zc[(size_t)c * NN + n];
  XT0[idx] = (__bf16)(z + BiasL[c]);
  XT1[idx] = (__bf16)(z + BiasL[NOPEN + c]);
}

__global__ __launch_bounds__(256) void k_leapfrog(const float* __restrict__ A0t,
                                                  const float* __restrict__ A1f,
                                                  const float* __restrict__ Zc,
                                                  float* __restrict__ Zold_inout) {
  const int idx = blockIdx.x * 256 + threadIdx.x;   // 128*1024
  const float ai = A0t[idx] + A1f[idx];
  Zold_inout[idx] = 2.0f * Zc[idx] - Zold_inout[idx] - H2 * ai;  // new Zc
}

__global__ __launch_bounds__(256) void k_close(const float* __restrict__ Kclose,
                                               const float* __restrict__ Zc,
                                               const float* __restrict__ Zold,
                                               float* __restrict__ out) {
  const int idx = blockIdx.x * 256 + threadIdx.x;   // 6144
  if (idx >= 2 * NCLOSE * NN) return;
  const int half = idx / (NCLOSE * NN);
  const int rem  = idx % (NCLOSE * NN);
  const int r = rem / NN, c = rem % NN;
  const float* Zp = half ? Zold : Zc;
  float acc = 0.f;
#pragma unroll 4
  for (int k = 0; k < NOPEN; ++k) acc += Kclose[r * NOPEN + k] * Zp[(size_t)k * NN + c];
  out[idx] = acc;
}

// ---------------------------------------------------------------------------
extern "C" void kernel_launch(void* const* d_in, const int* in_sizes, int n_in,
                              void* d_out, int out_size, void* d_ws, size_t ws_size,
                              hipStream_t stream) {
  const float* Z      = (const float*)d_in[0];
  const float* Kopen  = (const float*)d_in[1];
  const float* Kclose = (const float*)d_in[2];
  const float* W      = (const float*)d_in[3];
  const float* Bias   = (const float*)d_in[4];
  float* out = (float*)d_out;
  (void)ws_size; (void)n_in; (void)in_sizes; (void)out_size;

  size_t off = 0;
  auto carve = [&](size_t bytes) -> void* {
    off = (off + 255) & ~(size_t)255;
    void* p = (char*)d_ws + off;
    off += bytes;
    return p;
  };
  float*  ZcA   = (float*)carve(NOPEN * NN * 4);
  float*  ZcB   = (float*)carve(NOPEN * NN * 4);
  float*  G     = (float*)carve((size_t)NN * NN * 4);
  float*  D2    = (float*)carve((size_t)NN * NN * 4);     // becomes Wg
  float*  sq    = (float*)carve(NN * 4);
  float*  part  = (float*)carve(NN * 4);
  float*  sigma = (float*)carve(4);
  float*  deg   = (float*)carve(NN * 4);
  __bf16* Lb    = (__bf16*)carve((size_t)NN * NN * 2);    // symmetric
  __bf16* ZcT   = (__bf16*)carve((size_t)NN * NOPEN * 2);
  __bf16* XT0   = (__bf16*)carve((size_t)NN * NOPEN * 2);
  __bf16* XT1   = (__bf16*)carve((size_t)NN * NOPEN * 2);
  __bf16* Wf0   = (__bf16*)carve((size_t)9 * NHID * NOPEN * 2);
  __bf16* Wf1   = (__bf16*)carve((size_t)9 * NHID * NOPEN * 2);
  __bf16* Wt0   = (__bf16*)carve((size_t)9 * NHID * NOPEN * 2);
  __bf16* Wt1   = (__bf16*)carve((size_t)9 * NHID * NOPEN * 2);
  float*  Y0    = (float*)carve((size_t)NHID * NN * 4);
  __bf16* Y1b   = (__bf16*)carve((size_t)NHID * NN * 2);
  float*  Y1L   = (float*)carve((size_t)NHID * NN * 4);
  __bf16* A0nT  = (__bf16*)carve((size_t)NN * NHID * 2);
  __bf16* A1nT  = (__bf16*)carve((size_t)NN * NHID * 2);
  float*  A0t   = (float*)carve((size_t)NOPEN * NN * 4);
  __bf16* A1tb  = (__bf16*)carve((size_t)NOPEN * NN * 2);
  float*  A1f   = (float*)carve((size_t)NOPEN * NN * 4);

  float* Zc   = ZcA;
  float* Zold = ZcB;

  const dim3 w32(32);
  const int EW = (NOPEN * NN) / 256;

  k_open<<<EW, 256, 0, stream>>>(Kopen, Z, Zc, Zold);

  for (int i = 0; i < NLAYERS; ++i) {
    if (i % 10 == 0) {
      k_transp_zc<<<EW, 256, 0, stream>>>(Zc, ZcT);
      // G = ZcT * ZcT^T  (== Zc^T Zc), K = 128
      k_gemm_nt<<<dim3(NN / 32, NN / 32), w32, 0, stream>>>(
          ZcT, NOPEN, ZcT, NOPEN, G, NN, NOPEN);
      k_colsq<<<NN, 128, 0, stream>>>(Zc, sq);
      k_d2row<<<NN, 256, 0, stream>>>(G, sq, D2, part);
      k_sigma<<<1, 256, 0, stream>>>(part, sigma);
      k_wgdeg<<<NN, 256, 0, stream>>>(D2, sigma, deg);
      k_lap<<<(NN * NN) / 256, 256, 0, stream>>>(D2, deg, Lb);
    }
    const float* W0 = W + (size_t)(i * 2 + 0) * NHID * NOPEN * 9;
    const float* W1 = W + (size_t)(i * 2 + 1) * NHID * NOPEN * 9;
    const float* Bl = Bias + (size_t)i * 2 * NOPEN;

    k_repack<<<(9 * NHID * NOPEN + 255) / 256, 256, 0, stream>>>(
        W0, W1, Wf0, Wf1, Wt0, Wt1);
    k_biasadd_T<<<EW, 256, 0, stream>>>(Zc, Bl, XT0, XT1);

    // forward convs (128 -> 256 ch)
    k_conv_nt<<<dim3(NN / 32, NHID / 32), w32, 0, stream>>>(
        Wf0, XT0, Y0, (__bf16*)nullptr, NHID, NOPEN, 0);
    k_conv_nt<<<dim3(NN / 32, NHID / 32), w32, 0, stream>>>(
        Wf1, XT1, (float*)nullptr, Y1b, NHID, NOPEN, 0);

    // branch 0: instnorm+relu (transposed bf16 out for conv^T input)
    k_instnorm_relu_T<<<NHID, 256, 0, stream>>>(Y0, A0nT, NHID);

    // branch 1: (conv @ L) via NT GEMM (L symmetric), instnorm+relu
    k_gemm_nt<<<dim3(NN / 32, NHID / 32), w32, 0, stream>>>(
        Y1b, NN, Lb, NN, Y1L, NN, NN);
    k_instnorm_relu_T<<<NHID, 256, 0, stream>>>(Y1L, A1nT, NHID);

    // transpose convs (256 -> 128 ch)
    k_conv_nt<<<dim3(NN / 32, NOPEN / 32), w32, 0, stream>>>(
        Wt0, A0nT, A0t, (__bf16*)nullptr, NOPEN, NHID, 1);
    k_conv_nt<<<dim3(NN / 32, NOPEN / 32), w32, 0, stream>>>(
        Wt1, A1nT, (float*)nullptr, A1tb, NOPEN, NHID, 1);

    // branch 1: @ L^T == @ L (symmetric) via NT GEMM
    k_gemm_nt<<<dim3(NN / 32, NOPEN / 32), w32, 0, stream>>>(
        A1tb, NN, Lb, NN, A1f, NN, NN);

    k_leapfrog<<<EW, 256, 0, stream>>>(A0t, A1f, Zc, Zold);
    float* t = Zc; Zc = Zold; Zold = t;
  }

  k_close<<<(2 * NCLOSE * NN + 255) / 256, 256, 0, stream>>>(Kclose, Zc, Zold, out);
}
